// VoxelResBackBone8x_29669634081515
// MI455X (gfx1250) — matrix operand order repaired and checked by reading, not compiled, not run on hardware
//
#include <hip/hip_runtime.h>
#include <hip/hip_bf16.h>

typedef _Float16 v16h __attribute__((ext_vector_type(16)));
typedef _Float16 v8h  __attribute__((ext_vector_type(8)));
typedef _Float16 v4h  __attribute__((ext_vector_type(4)));
typedef float    v8f  __attribute__((ext_vector_type(8)));

#define BN_EPS 0.001f

// ---------------------------------------------------------------------------
// All activations live on HALO-PADDED grids: (D+2, H+2, W+2, C) channel-last
// f16, plus 2 zeroed guard planes at the tail.  Halos/guards are zeroed by
// memset each launch; convs write interiors only.  Tap reads need NO bounds
// checks: out-of-window taps read zeros (halo) or garbage annihilated by
// zero-padded weights.  ALL geometry is compile-time, so tap offsets fold
// into VGLOBAL IOFFSET immediates.
// ---------------------------------------------------------------------------
struct ConvArgs {
  const _Float16* __restrict__ x;        // padded input (DinP,HinP,WinP,CIN)
  const _Float16* __restrict__ wq;       // packed weights [Cout][KPAD], K=tap*CIN+ci
  const float* __restrict__ gamma;
  const float* __restrict__ beta;
  const float* __restrict__ mean;
  const float* __restrict__ var;
  const float* __restrict__ bias;        // may be null
  const float* __restrict__ maskOut;     // unpadded occupancy mask at OUTPUT res
  const _Float16* __restrict__ residual; // may be null; PADDED output-res grid, >=0
  _Float16* __restrict__ y;              // padded output act (may be null)
  float* __restrict__ yf32;              // final output NCDHW f32, unpadded (may be null)
};

// ---------------------------------------------------------------------------
// Implicit-GEMM 3D conv on V_WMMA_F32_16X16X32_F16.
// Wave tile: 16 x-consecutive output positions (one (z,y) row) x NT*16 chans.
// A-frag (ISA layout): lane row M=lane%16; K runs of 8 at (lane<16?0:8) and
// (lane<16?16:24) per 32-chunk.  B-frag: lane col N=lane%16, same K runs as A
// C/D: VGPR v holds M = v + (lane<16?0:8).
// ---------------------------------------------------------------------------
template<int CIN, int KD, int KH, int KW, int NT,
         int HinP, int WinP, int Dout, int Hout, int Wout,
         int HoutP, int WoutP, int Cout,
         int SZ, int SY, int SX, int AZ, int AY, int AX>
__global__ __launch_bounds__(256) void conv3d_wmma(ConvArgs p) {
  constexpr int NTAPS = KD * KH * KW;
  constexpr int KTOT  = NTAPS * CIN;
  constexpr int KPAD  = (KTOT + 31) & ~31;
  constexpr int KHW   = KH * KW;
  constexpr int XT    = (Wout + 15) / 16;
  constexpr int TILES = Dout * Hout * XT;
  constexpr int MOUT  = Dout * Hout * Wout;

  const int wave = threadIdx.x >> 5;
  const int lane = threadIdx.x & 31;
  const int tile = blockIdx.x * (blockDim.x >> 5) + wave;
  if (tile >= TILES) return;                         // wave-uniform exit

  const int xc = tile % XT;                          // constant divisors
  const int tt = tile / XT;
  const int y  = tt % Hout;
  const int z  = tt / Hout;

  const bool hi  = lane >= 16;
  const int  row = lane & 15;
  const int  ncb = blockIdx.y * (16 * NT) + row;     // first out-channel of tile 0

  // Gather base: uniform (z,y) part + per-lane x part; all reads in-bounds by
  // construction (halo + guard planes).
  const int izp = z * SZ + AZ;
  const int iyp = y * SY + AY;
  const int ixp = (xc * 16 + row) * SX + AX;
  const int base = ((izp * HinP + iyp) * WinP + ixp) * CIN;

  const _Float16* wb[NT];
#pragma unroll
  for (int t = 0; t < NT; ++t)
    wb[t] = p.wq + (size_t)(ncb + 16 * t) * KPAD + (hi ? 8 : 0);

  v8f acc[NT] = {};

  auto mm = [&](int k, v16h a) {
#pragma unroll
    for (int t = 0; t < NT; ++t) {
      v8h b0 = *(const v8h*)(wb[t] + k);
      v8h b1 = *(const v8h*)(wb[t] + k + 16);
      v16h b = __builtin_shufflevector(b0, b1, 0, 1, 2, 3, 4, 5, 6, 7,
                                       8, 9, 10, 11, 12, 13, 14, 15);
      acc[t] = __builtin_amdgcn_wmma_f32_16x16x32_f16(false, a, false, b,
                                                      (short)0, acc[t], false, false);
    }
  };

  if constexpr (CIN % 32 == 0) {
    const _Float16* bpA = p.x + base + (hi ? 8 : 0);
    const _Float16* bpB = p.x + base + (hi ? 24 : 16);
    for (int dz = 0; dz < KD; ++dz)
      for (int dy = 0; dy < KH; ++dy) {
        const int aoff  = (dz * HinP + dy) * WinP * CIN;   // 1 uniform add/iter
        const int kbase = (dz * KH + dy) * KW * CIN;
#pragma unroll
        for (int dx = 0; dx < KW; ++dx) {
#pragma unroll
          for (int cc = 0; cc < CIN / 32; ++cc) {
            const int o = dx * CIN + cc * 32;              // literal
            v8h r0 = *(const v8h*)(bpA + aoff + o);        // imm-offset b128
            v8h r1 = *(const v8h*)(bpB + aoff + o);
            v16h a = __builtin_shufflevector(r0, r1, 0, 1, 2, 3, 4, 5, 6, 7,
                                             8, 9, 10, 11, 12, 13, 14, 15);
            mm(kbase + o, a);
          }
        }
      }
  } else if constexpr (CIN == 16) {
    // One 32-chunk = two (wave-uniform) taps; all offsets are literals that
    // fold straight into the load's IOFFSET field.
    const _Float16* bpA = p.x + base + (hi ? 8 : 0);
    constexpr int CH = KPAD / 32;
#pragma unroll
    for (int c = 0; c < CH; ++c) {
      const int t0 = 2 * c, t1 = 2 * c + 1;
      const int o0 = (((t0 / KHW) * HinP + (t0 % KHW) / KW) * WinP + t0 % KW) * 16;
      const int o1 = (((t1 / KHW) * HinP + (t1 % KHW) / KW) * WinP + t1 % KW) * 16;
      v8h r0 = *(const v8h*)(bpA + o0);
      v8h r1 = *(const v8h*)(bpA + o1);
      v16h a = __builtin_shufflevector(r0, r1, 0, 1, 2, 3, 4, 5, 6, 7,
                                       8, 9, 10, 11, 12, 13, 14, 15);
      mm(c * 32, a);
    }
  } else {
    static_assert(CIN == 4, "unsupported CIN");
    // One 32-chunk = eight taps; lo lanes take taps 8c+{0,1,4,5}, hi lanes
    // 8c+{2,3,6,7}: one cndmask between two LITERAL offsets each.
    const _Float16* bp4 = p.x + base;
    auto toffC = [&](int t) {
      return (((t / KHW) * HinP + (t % KHW) / KW) * WinP + t % KW) * 4;
    };
    constexpr int CH = KPAD / 32;
#pragma unroll
    for (int c = 0; c < CH; ++c) {
      const int oA = hi ? toffC(8 * c + 2) : toffC(8 * c + 0);
      const int oB = hi ? toffC(8 * c + 3) : toffC(8 * c + 1);
      const int oC = hi ? toffC(8 * c + 6) : toffC(8 * c + 4);
      const int oD = hi ? toffC(8 * c + 7) : toffC(8 * c + 5);
      v4h q0 = *(const v4h*)(bp4 + oA);
      v4h q1 = *(const v4h*)(bp4 + oB);
      v4h q2 = *(const v4h*)(bp4 + oC);
      v4h q3 = *(const v4h*)(bp4 + oD);
      v8h r0 = __builtin_shufflevector(q0, q1, 0, 1, 2, 3, 4, 5, 6, 7);
      v8h r1 = __builtin_shufflevector(q2, q3, 0, 1, 2, 3, 4, 5, 6, 7);
      v16h a = __builtin_shufflevector(r0, r1, 0, 1, 2, 3, 4, 5, 6, 7,
                                       8, 9, 10, 11, 12, 13, 14, 15);
      mm(c * 32, a);
    }
  }

  // Epilogue: folded BN(+bias), mask, optional residual, ReLU.  All output
  // indices linear in x (no per-row decode); row bases use constant muls.
  float scale[NT], shift[NT];
#pragma unroll
  for (int t = 0; t < NT; ++t) {
    const int n = ncb + 16 * t;
    const float g = p.gamma[n], bt = p.beta[n], mn = p.mean[n], vr = p.var[n];
    const float bs = p.bias ? p.bias[n] : 0.0f;
    scale[t] = g * rsqrtf(vr + BN_EPS);
    shift[t] = (bs - mn) * scale[t] + bt;
  }

  const int orow = ((z + 1) * HoutP + (y + 1)) * WoutP + 1;  // padded row base
  const int mrow = (z * Hout + y) * Wout;                    // unpadded row base

#pragma unroll
  for (int v = 0; v < 8; ++v) {
    const int xr = xc * 16 + v + (hi ? 8 : 0);
    // Guard only needed when Wout is not a multiple of the 16-wide x-tile.
    if ((Wout % 16 == 0) || (xr < Wout)) {
      const float mk = p.maskOut[mrow + xr];
      const int pcell = orow + xr;
#pragma unroll
      for (int t = 0; t < NT; ++t) {
        const int n = ncb + 16 * t;
        const float val = acc[t][v] * scale[t] + shift[t];
        float outv;
        if (p.residual) {
          const float res  = (float)p.residual[(size_t)pcell * Cout + n];
          const float ymsk = (mk != 0.0f) ? val : 0.0f;
          outv = fmaxf(ymsk + res, 0.0f);
        } else {
          outv = (mk != 0.0f) ? fmaxf(val, 0.0f) : 0.0f;
        }
        if (p.y)    p.y[(size_t)pcell * Cout + n] = (_Float16)outv;
        if (p.yf32) p.yf32[(size_t)n * MOUT + mrow + xr] = outv;   // NCDHW, N=1
      }
    }
  }
}

// ---------------------------------------------------------------------------
// Scatter voxels into padded channel-last f16 grid + unpadded occupancy mask
// ---------------------------------------------------------------------------
__global__ void scatter_kernel(const float* __restrict__ feats,
                               const int* __restrict__ coords,
                               _Float16* __restrict__ act, float* __restrict__ m1,
                               int nvox, int H, int W, int Hp, int Wp, int C) {
  const int v = blockIdx.x * blockDim.x + threadIdx.x;
  if (v >= nvox) return;
  const int z = coords[v * 4 + 1];
  const int y = coords[v * 4 + 2];
  const int x = coords[v * 4 + 3];
  const size_t cell = ((size_t)(z + 1) * Hp + (y + 1)) * Wp + (x + 1);
  for (int c = 0; c < C; ++c) act[cell * C + c] = (_Float16)feats[v * C + c];
  m1[((size_t)z * H + y) * W + x] = 1.0f;
}

// ---------------------------------------------------------------------------
// Mask max-pool (reduce_window max, init 0) on unpadded masks
// ---------------------------------------------------------------------------
__global__ void maskpool_kernel(const float* __restrict__ in, float* __restrict__ out,
                                int Din, int Hin, int Win, int Dout, int Hout, int Wout,
                                int kd, int kh, int kw, int sz, int sy, int sx,
                                int pz, int py, int px) {
  const int i = blockIdx.x * blockDim.x + threadIdx.x;
  const int M = Dout * Hout * Wout;
  if (i >= M) return;
  const int ox = i % Wout; int t = i / Wout; const int oy = t % Hout; const int oz = t / Hout;
  float mx = 0.0f;
  for (int dz = 0; dz < kd; ++dz) {
    const int iz = oz * sz - pz + dz; if ((unsigned)iz >= (unsigned)Din) continue;
    for (int dy = 0; dy < kh; ++dy) {
      const int iy = oy * sy - py + dy; if ((unsigned)iy >= (unsigned)Hin) continue;
      for (int dx = 0; dx < kw; ++dx) {
        const int ix = ox * sx - px + dx; if ((unsigned)ix >= (unsigned)Win) continue;
        mx = fmaxf(mx, in[((size_t)iz * Hin + iy) * Win + ix]);
      }
    }
  }
  out[i] = mx;
}

// ---------------------------------------------------------------------------
// Pack f32 weights (O,I,kd,kh,kw) -> f16 [Cout][Kpad], K = tap*Cin + ci,
// zero-padded to Kpad (garbage A data at padded K is annihilated here).
// ---------------------------------------------------------------------------
__global__ void prep_weights(const float* __restrict__ w, _Float16* __restrict__ wq,
                             int Cout, int Cin, int ntaps, int Kpad) {
  const int i = blockIdx.x * blockDim.x + threadIdx.x;
  const int total = Cout * Kpad;
  if (i >= total) return;
  const int o = i / Kpad; const int k = i - o * Kpad;
  float v = 0.0f;
  const int Ktot = Cin * ntaps;
  if (k < Ktot) {
    const int tap = k / Cin; const int ci = k - tap * Cin;
    v = w[((size_t)o * Cin + ci) * ntaps + tap];
  }
  wq[i] = (_Float16)v;
}

// ---------------------------------------------------------------------------
// Host side
// ---------------------------------------------------------------------------
template<int CIN, int KD, int KH, int KW, int NT,
         int HinP, int WinP, int Dout, int Hout, int Wout,
         int HoutP, int WoutP, int Cout,
         int SZ, int SY, int SX, int PZ, int PY, int PX>
static void launch_conv(hipStream_t stream, const ConvArgs& a) {
  constexpr int XT = (Wout + 15) / 16;
  constexpr int tiles = Dout * Hout * XT;
  dim3 grid((tiles + 7) / 8, Cout / (16 * NT));
  conv3d_wmma<CIN, KD, KH, KW, NT, HinP, WinP, Dout, Hout, Wout, HoutP, WoutP,
              Cout, SZ, SY, SX, 1 - PZ, 1 - PY, 1 - PX><<<grid, 256, 0, stream>>>(a);
}

static ConvArgs mkArgs(const _Float16* x, const _Float16* wq,
                       const float* beta, const float* gamma,
                       const float* mean, const float* var, const float* bias,
                       const float* mask, const _Float16* res,
                       _Float16* y, float* yf32) {
  ConvArgs a{};
  a.x = x; a.wq = wq;
  a.gamma = gamma; a.beta = beta; a.mean = mean; a.var = var; a.bias = bias;
  a.maskOut = mask; a.residual = res; a.y = y; a.yf32 = yf32;
  return a;
}

extern "C" void kernel_launch(void* const* d_in, const int* in_sizes, int n_in,
                              void* d_out, int out_size, void* d_ws, size_t ws_size,
                              hipStream_t stream) {
  (void)in_sizes; (void)n_in; (void)out_size; (void)ws_size;

  // Input index table (jax pytree flatten: top dict insertion order,
  // nested dicts key-sorted).  Block: b1,b2,bn1{beta,gamma,mean,var},
  // bn2{...},w1,w2.  Down/in/out: bn{beta,gamma,mean,var},w.
  enum { FEATS = 0, COORDS = 1,
         C1B1 = 2, C1B2 = 14, C2B1 = 26, C2B2 = 38, C2D = 50,
         C3B1 = 55, C3B2 = 67, C3D = 79, C4B1 = 84, C4B2 = 96, C4D = 108,
         INL = 113, OUTL = 118 };
  auto F = [&](int i) { return (const float*)d_in[i]; };

  // Unpadded / padded stage geometry.
  const int D1 = 25, H1 = 160, W1 = 160, D1p = 27, H1p = 162, W1p = 162;
  const int D2 = 13, H2 = 80,  W2 = 80,  D2p = 15, H2p = 82,  W2p = 82;
  const int D3 = 7,  H3 = 40,  W3 = 40,  D3p = 9,  H3p = 42,  W3p = 42;
  const int D4 = 3,  H4 = 20,  W4 = 20,  D4p = 5,  H4p = 22,  W4p = 22;
  const int D5 = 1,  H5 = 20,  W5 = 20;
  const size_t M1 = (size_t)D1 * H1 * W1;
  const size_t M2 = (size_t)D2 * H2 * W2;
  const size_t M3 = (size_t)D3 * H3 * W3;
  const size_t M4 = (size_t)D4 * H4 * W4;
  const size_t M5 = (size_t)D5 * H5 * W5;
  const int NVOX = 80000, CIN0 = 4;

  // ---- workspace bump allocator ----
  char* base = (char*)d_ws; size_t off = 0;
  auto alloc = [&](size_t bytes) -> void* {
    off = (off + 255) & ~(size_t)255;
    void* p = base + off; off += bytes; return p;
  };
  size_t actBytes[8];
  auto actAlloc = [&](int Dp, int Hp, int Wp, int C, size_t* sz) -> _Float16* {
    size_t bytes = ((size_t)Dp * Hp * Wp + 2 * (size_t)Hp * Wp) * C * 2;
    if (sz) *sz = bytes;
    return (_Float16*)alloc(bytes);
  };

  size_t inBytes;
  _Float16* actIn = actAlloc(D1p, H1p, W1p, CIN0, &inBytes);
  float* m1b = (float*)alloc(M1 * 4);
  float* m2b = (float*)alloc(M2 * 4);
  float* m3b = (float*)alloc(M3 * 4);
  float* m4b = (float*)alloc(M4 * 4);
  float* m5b = (float*)alloc(M5 * 4);
  _Float16* a1 = actAlloc(D1p, H1p, W1p, 16, &actBytes[0]);
  _Float16* t1 = actAlloc(D1p, H1p, W1p, 16, &actBytes[1]);
  _Float16* a2 = actAlloc(D2p, H2p, W2p, 32, &actBytes[2]);
  _Float16* t2 = actAlloc(D2p, H2p, W2p, 32, &actBytes[3]);
  _Float16* a3 = actAlloc(D3p, H3p, W3p, 64, &actBytes[4]);
  _Float16* t3 = actAlloc(D3p, H3p, W3p, 64, &actBytes[5]);
  _Float16* a4 = actAlloc(D4p, H4p, W4p, 128, &actBytes[6]);
  _Float16* t4 = actAlloc(D4p, H4p, W4p, 128, &actBytes[7]);

  // ---- zero halos/guards + scatter ----
  hipMemsetAsync(actIn, 0, inBytes, stream);
  hipMemsetAsync(m1b, 0, M1 * 4, stream);
  hipMemsetAsync(a1, 0, actBytes[0], stream);
  hipMemsetAsync(t1, 0, actBytes[1], stream);
  hipMemsetAsync(a2, 0, actBytes[2], stream);
  hipMemsetAsync(t2, 0, actBytes[3], stream);
  hipMemsetAsync(a3, 0, actBytes[4], stream);
  hipMemsetAsync(t3, 0, actBytes[5], stream);
  hipMemsetAsync(a4, 0, actBytes[6], stream);
  hipMemsetAsync(t4, 0, actBytes[7], stream);
  scatter_kernel<<<(NVOX + 255) / 256, 256, 0, stream>>>(
      F(FEATS), (const int*)d_in[COORDS], actIn, m1b, NVOX, H1, W1, H1p, W1p, CIN0);

  // ---- weight packing ----
  auto prep = [&](int idx, int Cout, int Cin, int ntaps) -> const _Float16* {
    const int Ktot = Cin * ntaps;
    const int Kpad = (Ktot + 31) & ~31;
    _Float16* wq = (_Float16*)alloc((size_t)Cout * Kpad * 2);
    const int total = Cout * Kpad;
    prep_weights<<<(total + 255) / 256, 256, 0, stream>>>(F(idx), wq, Cout, Cin, ntaps, Kpad);
    return wq;
  };
  const _Float16* wIn  = prep(INL + 4, 16, 4, 27);
  const _Float16* w11a = prep(C1B1 + 10, 16, 16, 27);
  const _Float16* w11b = prep(C1B1 + 11, 16, 16, 27);
  const _Float16* w12a = prep(C1B2 + 10, 16, 16, 27);
  const _Float16* w12b = prep(C1B2 + 11, 16, 16, 27);
  const _Float16* w2d  = prep(C2D + 4, 32, 16, 27);
  const _Float16* w21a = prep(C2B1 + 10, 32, 32, 27);
  const _Float16* w21b = prep(C2B1 + 11, 32, 32, 27);
  const _Float16* w22a = prep(C2B2 + 10, 32, 32, 27);
  const _Float16* w22b = prep(C2B2 + 11, 32, 32, 27);
  const _Float16* w3d  = prep(C3D + 4, 64, 32, 27);
  const _Float16* w31a = prep(C3B1 + 10, 64, 64, 27);
  const _Float16* w31b = prep(C3B1 + 11, 64, 64, 27);
  const _Float16* w32a = prep(C3B2 + 10, 64, 64, 27);
  const _Float16* w32b = prep(C3B2 + 11, 64, 64, 27);
  const _Float16* w4d  = prep(C4D + 4, 128, 64, 27);
  const _Float16* w41a = prep(C4B1 + 10, 128, 128, 27);
  const _Float16* w41b = prep(C4B1 + 11, 128, 128, 27);
  const _Float16* w42a = prep(C4B2 + 10, 128, 128, 27);
  const _Float16* w42b = prep(C4B2 + 11, 128, 128, 27);
  const _Float16* wOut = prep(OUTL + 4, 128, 128, 3);

  // ---- mask pooling chain ----
  auto pool = [&](const float* in, float* out, int Di, int Hi, int Wi,
                  int Do, int Ho, int Wo, int kd, int kh, int kw,
                  int sz, int sy, int sx, int pz, int py, int px) {
    const int M = Do * Ho * Wo;
    maskpool_kernel<<<(M + 255) / 256, 256, 0, stream>>>(
        in, out, Di, Hi, Wi, Do, Ho, Wo, kd, kh, kw, sz, sy, sx, pz, py, px);
  };
  pool(m1b, m2b, D1, H1, W1, D2, H2, W2, 3, 3, 3, 2, 2, 2, 1, 1, 1);
  pool(m2b, m3b, D2, H2, W2, D3, H3, W3, 3, 3, 3, 2, 2, 2, 1, 1, 1);
  pool(m3b, m4b, D3, H3, W3, D4, H4, W4, 3, 3, 3, 2, 2, 2, 0, 1, 1);
  pool(m4b, m5b, D4, H4, W4, D5, H5, W5, 3, 1, 1, 2, 1, 1, 0, 0, 0);

  // ---- network (geometry in template args) ----
  // stem: Cin=4 -> 16 @ s1
  launch_conv<4, 3, 3, 3, 1, 162, 162, 25, 160, 160, 162, 162, 16, 1, 1, 1, 1, 1, 1>(
      stream, mkArgs(actIn, wIn, F(INL), F(INL + 1), F(INL + 2), F(INL + 3),
                     nullptr, m1b, nullptr, a1, nullptr));
  // stage 1 (16 ch @ s1)
  launch_conv<16, 3, 3, 3, 1, 162, 162, 25, 160, 160, 162, 162, 16, 1, 1, 1, 1, 1, 1>(
      stream, mkArgs(a1, w11a, F(C1B1 + 2), F(C1B1 + 3), F(C1B1 + 4), F(C1B1 + 5),
                     F(C1B1 + 0), m1b, nullptr, t1, nullptr));
  launch_conv<16, 3, 3, 3, 1, 162, 162, 25, 160, 160, 162, 162, 16, 1, 1, 1, 1, 1, 1>(
      stream, mkArgs(t1, w11b, F(C1B1 + 6), F(C1B1 + 7), F(C1B1 + 8), F(C1B1 + 9),
                     F(C1B1 + 1), m1b, a1, a1, nullptr));
  launch_conv<16, 3, 3, 3, 1, 162, 162, 25, 160, 160, 162, 162, 16, 1, 1, 1, 1, 1, 1>(
      stream, mkArgs(a1, w12a, F(C1B2 + 2), F(C1B2 + 3), F(C1B2 + 4), F(C1B2 + 5),
                     F(C1B2 + 0), m1b, nullptr, t1, nullptr));
  launch_conv<16, 3, 3, 3, 1, 162, 162, 25, 160, 160, 162, 162, 16, 1, 1, 1, 1, 1, 1>(
      stream, mkArgs(t1, w12b, F(C1B2 + 6), F(C1B2 + 7), F(C1B2 + 8), F(C1B2 + 9),
                     F(C1B2 + 1), m1b, a1, a1, nullptr));
  // down 2 (16 -> 32, stride 2, s1 -> s2) + stage 2
  launch_conv<16, 3, 3, 3, 2, 162, 162, 13, 80, 80, 82, 82, 32, 2, 2, 2, 1, 1, 1>(
      stream, mkArgs(a1, w2d, F(C2D), F(C2D + 1), F(C2D + 2), F(C2D + 3),
                     nullptr, m2b, nullptr, a2, nullptr));
  launch_conv<32, 3, 3, 3, 2, 82, 82, 13, 80, 80, 82, 82, 32, 1, 1, 1, 1, 1, 1>(
      stream, mkArgs(a2, w21a, F(C2B1 + 2), F(C2B1 + 3), F(C2B1 + 4), F(C2B1 + 5),
                     F(C2B1 + 0), m2b, nullptr, t2, nullptr));
  launch_conv<32, 3, 3, 3, 2, 82, 82, 13, 80, 80, 82, 82, 32, 1, 1, 1, 1, 1, 1>(
      stream, mkArgs(t2, w21b, F(C2B1 + 6), F(C2B1 + 7), F(C2B1 + 8), F(C2B1 + 9),
                     F(C2B1 + 1), m2b, a2, a2, nullptr));
  launch_conv<32, 3, 3, 3, 2, 82, 82, 13, 80, 80, 82, 82, 32, 1, 1, 1, 1, 1, 1>(
      stream, mkArgs(a2, w22a, F(C2B2 + 2), F(C2B2 + 3), F(C2B2 + 4), F(C2B2 + 5),
                     F(C2B2 + 0), m2b, nullptr, t2, nullptr));
  launch_conv<32, 3, 3, 3, 2, 82, 82, 13, 80, 80, 82, 82, 32, 1, 1, 1, 1, 1, 1>(
      stream, mkArgs(t2, w22b, F(C2B2 + 6), F(C2B2 + 7), F(C2B2 + 8), F(C2B2 + 9),
                     F(C2B2 + 1), m2b, a2, a2, nullptr));
  // down 3 (32 -> 64, s2 -> s3) + stage 3
  launch_conv<32, 3, 3, 3, 2, 82, 82, 7, 40, 40, 42, 42, 64, 2, 2, 2, 1, 1, 1>(
      stream, mkArgs(a2, w3d, F(C3D), F(C3D + 1), F(C3D + 2), F(C3D + 3),
                     nullptr, m3b, nullptr, a3, nullptr));
  launch_conv<64, 3, 3, 3, 2, 42, 42, 7, 40, 40, 42, 42, 64, 1, 1, 1, 1, 1, 1>(
      stream, mkArgs(a3, w31a, F(C3B1 + 2), F(C3B1 + 3), F(C3B1 + 4), F(C3B1 + 5),
                     F(C3B1 + 0), m3b, nullptr, t3, nullptr));
  launch_conv<64, 3, 3, 3, 2, 42, 42, 7, 40, 40, 42, 42, 64, 1, 1, 1, 1, 1, 1>(
      stream, mkArgs(t3, w31b, F(C3B1 + 6), F(C3B1 + 7), F(C3B1 + 8), F(C3B1 + 9),
                     F(C3B1 + 1), m3b, a3, a3, nullptr));
  launch_conv<64, 3, 3, 3, 2, 42, 42, 7, 40, 40, 42, 42, 64, 1, 1, 1, 1, 1, 1>(
      stream, mkArgs(a3, w32a, F(C3B2 + 2), F(C3B2 + 3), F(C3B2 + 4), F(C3B2 + 5),
                     F(C3B2 + 0), m3b, nullptr, t3, nullptr));
  launch_conv<64, 3, 3, 3, 2, 42, 42, 7, 40, 40, 42, 42, 64, 1, 1, 1, 1, 1, 1>(
      stream, mkArgs(t3, w32b, F(C3B2 + 6), F(C3B2 + 7), F(C3B2 + 8), F(C3B2 + 9),
                     F(C3B2 + 1), m3b, a3, a3, nullptr));
  // down 4 (64 -> 128, pad (0,1,1), s3 -> s4) + stage 4
  launch_conv<64, 3, 3, 3, 2, 42, 42, 3, 20, 20, 22, 22, 128, 2, 2, 2, 0, 1, 1>(
      stream, mkArgs(a3, w4d, F(C4D), F(C4D + 1), F(C4D + 2), F(C4D + 3),
                     nullptr, m4b, nullptr, a4, nullptr));
  launch_conv<128, 3, 3, 3, 2, 22, 22, 3, 20, 20, 22, 22, 128, 1, 1, 1, 1, 1, 1>(
      stream, mkArgs(a4, w41a, F(C4B1 + 2), F(C4B1 + 3), F(C4B1 + 4), F(C4B1 + 5),
                     F(C4B1 + 0), m4b, nullptr, t4, nullptr));
  launch_conv<128, 3, 3, 3, 2, 22, 22, 3, 20, 20, 22, 22, 128, 1, 1, 1, 1, 1, 1>(
      stream, mkArgs(t4, w41b, F(C4B1 + 6), F(C4B1 + 7), F(C4B1 + 8), F(C4B1 + 9),
                     F(C4B1 + 1), m4b, a4, a4, nullptr));
  launch_conv<128, 3, 3, 3, 2, 22, 22, 3, 20, 20, 22, 22, 128, 1, 1, 1, 1, 1, 1>(
      stream, mkArgs(a4, w42a, F(C4B2 + 2), F(C4B2 + 3), F(C4B2 + 4), F(C4B2 + 5),
                     F(C4B2 + 0), m4b, nullptr, t4, nullptr));
  launch_conv<128, 3, 3, 3, 2, 22, 22, 3, 20, 20, 22, 22, 128, 1, 1, 1, 1, 1, 1>(
      stream, mkArgs(t4, w42b, F(C4B2 + 6), F(C4B2 + 7), F(C4B2 + 8), F(C4B2 + 9),
                     F(C4B2 + 1), m4b, a4, a4, nullptr));
  // output conv (3,1,1) stride (2,1,1) pad 0 -> d_out (f32, NCDHW)
  launch_conv<128, 3, 1, 1, 2, 22, 22, 1, 20, 20, 22, 22, 128, 2, 1, 1, 0, 0, 0>(
      stream, mkArgs(a4, wOut, F(OUTL), F(OUTL + 1), F(OUTL + 2), F(OUTL + 3),
                     nullptr, m5b, nullptr, nullptr, (float*)d_out));
}
